// Flashback_17394617549205
// MI455X (gfx1250) — compile-verified, matrix operand order checked by reading, not run on hardware
//
#include <hip/hip_runtime.h>
#include <hip/hip_bf16.h>
#include <math.h>

typedef __attribute__((ext_vector_type(16))) _Float16 v16h;
typedef __attribute__((ext_vector_type(8)))  _Float16 v8h;
typedef __attribute__((ext_vector_type(8)))  float    v8f;
typedef __attribute__((ext_vector_type(4)))  float    v4f;

#define S_LEN 128
#define U_CNT 256
#define HID   128
#define VOC   8000
#define K2H   256
#define BPAD  264   // padded f16 row stride for LDS B tile (528B = 132 dwords = +4 banks/row)
#define LAM_T 0.1f
#define LAM_S 100.0f
#define DAYF  86400.0f
#define TWO_PI_OVER_DAY (6.28318530717958647692f / 86400.0f)

// ---- WMMA 16-bit A/B fragment builders (CDNA5 §7.12.2 layout) ----
// lane L: l = L%16 (row for A / col for B), g = L/16.
// element e: K = kbase + (e<8 ? 8g+e : 16+8g+(e-8))
__device__ __forceinline__ v16h frag_from_f32(const float* __restrict__ row, int kbase, int g) {
    const int k0 = kbase + 8 * g;
    v4f a0 = *(const v4f*)(row + k0);
    v4f a1 = *(const v4f*)(row + k0 + 4);
    v4f b0 = *(const v4f*)(row + k0 + 16);
    v4f b1 = *(const v4f*)(row + k0 + 20);
    v16h f;
#pragma unroll
    for (int e = 0; e < 4; ++e) {
        f[e]      = (_Float16)a0[e];
        f[e + 4]  = (_Float16)a1[e];
        f[e + 8]  = (_Float16)b0[e];
        f[e + 12] = (_Float16)b1[e];
    }
    return f;
}

__device__ __forceinline__ v16h frag_from_f16(const _Float16* row, int kbase, int g) {
    const int k0 = kbase + 8 * g;
    v8h lo = *(const v8h*)(row + k0);        // K = kbase+8g .. +7   (16B aligned)
    v8h hi = *(const v8h*)(row + k0 + 16);   // K = kbase+16+8g .. +7
    v16h f;
#pragma unroll
    for (int e = 0; e < 8; ++e) { f[e] = lo[e]; f[8 + e] = hi[e]; }
    return f;
}

// ---- kernel 1: fc_W f32 -> f16 ----
__global__ void cvt_f16_kernel(const float* __restrict__ in, _Float16* __restrict__ out, int n) {
    int i = blockIdx.x * blockDim.x + threadIdx.x;
    if (i < n) out[i] = (_Float16)in[i];
}

// ---- kernel 2: uls[j,u] = exp(-||p_u[u] - enc_emb[x[j,u]]||) ----
__global__ void __launch_bounds__(256) uls_kernel(
    const int* __restrict__ x, const int* __restrict__ active_user,
    const float* __restrict__ enc_emb, const float* __restrict__ user_emb,
    float* __restrict__ uls) {
    const int pair = blockIdx.x * 8 + (threadIdx.x >> 5);   // one wave per (j,u)
    const int lane = threadIdx.x & 31;
    const int u = pair & (U_CNT - 1);
    const int j = pair >> 8;                                // pair = j*U + u
    const float* pe = user_emb + (size_t)active_user[u] * HID;
    const float* xe = enc_emb  + (size_t)x[(size_t)j * U_CNT + u] * HID;
    float acc = 0.f;
#pragma unroll
    for (int k = lane; k < HID; k += 32) { float d = pe[k] - xe[k]; acc += d * d; }
#pragma unroll
    for (int off = 16; off >= 1; off >>= 1) acc += __shfl_xor(acc, off, 32);
    if (lane == 0) uls[pair] = expf(-sqrtf(acc));
}

// ---- kernel 3: tanh-RNN scan, 16 users/block, WMMA with weights in VGPRs ----
// Concatenated-K view: A = [x_emb(16x128) | h(16x128)], B[k][n] = (k<128 ? W_ih[n][k] : W_hh[n][k-128])
__global__ void __launch_bounds__(128) rnn_kernel(
    const int* __restrict__ x, const float* __restrict__ h0,
    const float* __restrict__ enc_emb,
    const float* __restrict__ W_ih, const float* __restrict__ W_hh,
    const float* __restrict__ b_rnn, float* __restrict__ out) {
    __shared__ _Float16 h_stage[16 * HID];
    const int tid  = threadIdx.x;
    const int wave = tid >> 5, lane = tid & 31;
    const int l16  = lane & 15, g = lane >> 4;
    const int u0   = blockIdx.x * 16;

    // preload all B fragments (weights) into registers: 8 K-steps x 2 N-tiles per wave
    v16h bfrag[8][2];
#pragma unroll
    for (int t2 = 0; t2 < 2; ++t2) {
        const int n = wave * 32 + t2 * 16 + l16;
#pragma unroll
        for (int kb = 0; kb < 4; ++kb)
            bfrag[kb][t2] = frag_from_f32(W_ih + (size_t)n * HID, kb * 32, g);
#pragma unroll
        for (int kb = 4; kb < 8; ++kb)
            bfrag[kb][t2] = frag_from_f32(W_hh + (size_t)n * HID, (kb - 4) * 32, g);
    }
    const float bias0 = b_rnn[wave * 32 + l16];
    const float bias1 = b_rnn[wave * 32 + 16 + l16];

    // init h stage from h0
    for (int idx = tid; idx < 16 * HID; idx += 128) {
        int m = idx >> 7, n = idx & (HID - 1);
        h_stage[idx] = (_Float16)h0[(size_t)(u0 + m) * HID + n];
    }
    __syncthreads();

    for (int i = 0; i < S_LEN; ++i) {
        const int xid = x[(size_t)i * U_CNT + u0 + l16];
        const float* xrow = enc_emb + (size_t)xid * HID;
        v8f acc0 = {}, acc1 = {};
#pragma unroll
        for (int kb = 0; kb < 8; ++kb) {
            v16h a = (kb < 4) ? frag_from_f32(xrow, kb * 32, g)
                              : frag_from_f16(h_stage + l16 * HID, (kb - 4) * 32, g);
            acc0 = __builtin_amdgcn_wmma_f32_16x16x32_f16(false, a, false, bfrag[kb][0],
                                                          (short)0, acc0, false, false);
            acc1 = __builtin_amdgcn_wmma_f32_16x16x32_f16(false, a, false, bfrag[kb][1],
                                                          (short)0, acc1, false, false);
        }
        __syncthreads();   // all reads of previous h complete
#pragma unroll
        for (int r = 0; r < 8; ++r) {
            const int m  = r + 8 * g;
            const int n0 = wave * 32 + l16;
            const float v0 = tanhf(acc0[r] + bias0);
            const float v1 = tanhf(acc1[r] + bias1);
            h_stage[m * HID + n0]      = (_Float16)v0;
            h_stage[m * HID + n0 + 16] = (_Float16)v1;
            const size_t base = ((size_t)i * U_CNT + u0 + m) * HID;
            out[base + n0]      = v0;
            out[base + n0 + 16] = v1;
        }
        __syncthreads();   // h for next step visible
    }
}

// ---- kernel 4: causal decay-weighted average, builds concatenated f16 A matrix ----
__global__ void __launch_bounds__(128) wavg_kernel(
    const float* __restrict__ t, const float* __restrict__ s,
    const int* __restrict__ active_user, const float* __restrict__ user_emb,
    const float* __restrict__ uls, const float* __restrict__ out_rnn,
    _Float16* __restrict__ A_h) {
    __shared__ float w_lds[S_LEN];
    const int i = blockIdx.x >> 8;          // blockIdx.x = i*U + u
    const int u = blockIdx.x & (U_CNT - 1);
    const int n = threadIdx.x;
    {
        const int j = n;                    // S_LEN == blockDim.x
        float w = 0.f;
        if (j <= i) {
            const float dt = t[(size_t)i * U_CNT + u] - t[(size_t)j * U_CNT + u];
            const float dx = s[((size_t)i * U_CNT + u) * 2 + 0] - s[((size_t)j * U_CNT + u) * 2 + 0];
            const float dy = s[((size_t)i * U_CNT + u) * 2 + 1] - s[((size_t)j * U_CNT + u) * 2 + 1];
            const float ds = sqrtf(dx * dx + dy * dy);
            const float ft = (cosf(dt * TWO_PI_OVER_DAY) + 1.f) * 0.5f * expf(-(dt / DAYF) * LAM_T);
            const float fs = expf(-ds * LAM_S);
            w = ft * fs * uls[(size_t)j * U_CNT + u] + 1e-10f;
        }
        w_lds[j] = w;
    }
    __syncthreads();
    float acc = 0.f, wsum = 0.f;
    for (int j = 0; j <= i; ++j) {
        const float w = w_lds[j];
        wsum += w;
        acc  += w * out_rnn[((size_t)j * U_CNT + u) * HID + n];
    }
    const size_t row = (size_t)i * U_CNT + u;
    A_h[row * K2H + n]       = (_Float16)(acc / wsum);
    A_h[row * K2H + HID + n] = (_Float16)user_emb[(size_t)active_user[u] * HID + n];
}

// ---- kernel 5: FC GEMM (32768 x 8000, K=256), WMMA, 128x128 block tile / 8 waves ----
// B tile staged into LDS once per block via gfx1250 async DMA (global_load_async_to_lds_b128),
// fenced with s_wait_asynccnt + workgroup barrier; fragments then built from LDS (ds_load_b128).
__global__ void __launch_bounds__(256) fc_kernel(
    const _Float16* __restrict__ A, const _Float16* __restrict__ B,  // B = fc_W f16 (V,256)
    const float* __restrict__ fc_b, float* __restrict__ y) {
    __shared__ _Float16 b_lds[128 * BPAD];
    const int tid  = threadIdx.x;
    const int wave = tid >> 5, lane = tid & 31;
    const int l16  = lane & 15, g = lane >> 4;
    const int m0   = blockIdx.x * 128 + wave * 16;   // wave's 16 output rows
    const int n0   = blockIdx.y * 128;               // block's 128 output cols

    // ---- async-stage 128 rows x 256 f16 of B into LDS: 4096 x 16B chunks, 16 per thread ----
#pragma unroll
    for (int it = 0; it < 16; ++it) {
        const int task = tid + it * 256;
        const int row  = task >> 5;                  // 0..127  (local col of output tile)
        const int chk  = task & 31;                  // 16B chunk within the 512B row
        int nr = n0 + row; if (nr >= VOC) nr = VOC - 1;   // clamp: OOB tiles never stored
        const _Float16* gsrc = B + (size_t)nr * K2H + chk * 8;
        const unsigned  ldst = (unsigned)(uintptr_t)(&b_lds[row * BPAD + chk * 8]);
        asm volatile("global_load_async_to_lds_b128 %0, %1, off"
                     :: "v"(ldst), "v"((unsigned long long)(uintptr_t)gsrc)
                     : "memory");
    }
    asm volatile("s_wait_asynccnt 0x0" ::: "memory");
    __syncthreads();

    const _Float16* arow = A + (size_t)(m0 + l16) * K2H;
    v8f acc[8];
    v8f zero = {};
#pragma unroll
    for (int t2 = 0; t2 < 8; ++t2) acc[t2] = zero;

#pragma unroll
    for (int kb = 0; kb < 8; ++kb) {
        const v16h a = frag_from_f16(arow, kb * 32, g);
#pragma unroll
        for (int t2 = 0; t2 < 8; ++t2) {
            const v16h b = frag_from_f16(&b_lds[(t2 * 16 + l16) * BPAD], kb * 32, g);
            acc[t2] = __builtin_amdgcn_wmma_f32_16x16x32_f16(false, a, false, b,
                                                             (short)0, acc[t2], false, false);
        }
    }
#pragma unroll
    for (int t2 = 0; t2 < 8; ++t2) {
        const int n = n0 + t2 * 16 + l16;
        if (n < VOC) {                                // uniform per 16-wide tile
            const float bias = fc_b[n];
#pragma unroll
            for (int r = 0; r < 8; ++r) {
                const int m = m0 + r + 8 * g;
                y[(size_t)m * VOC + n] = acc[t2][r] + bias;
            }
        }
    }
}

extern "C" void kernel_launch(void* const* d_in, const int* in_sizes, int n_in,
                              void* d_out, int out_size, void* d_ws, size_t ws_size,
                              hipStream_t stream) {
    const int*   x        = (const int*)  d_in[0];
    const float* t        = (const float*)d_in[1];
    const float* s        = (const float*)d_in[2];
    const float* h0       = (const float*)d_in[3];
    const int*   active   = (const int*)  d_in[4];
    const float* enc_emb  = (const float*)d_in[5];
    const float* user_emb = (const float*)d_in[6];
    const float* W_ih     = (const float*)d_in[7];
    const float* W_hh     = (const float*)d_in[8];
    const float* b_rnn    = (const float*)d_in[9];
    const float* fc_W     = (const float*)d_in[10];
    const float* fc_b     = (const float*)d_in[11];
    float* y = (float*)d_out;

    char* ws = (char*)d_ws;
    float*    out_rnn = (float*)ws;     ws += (size_t)S_LEN * U_CNT * HID * sizeof(float);
    float*    uls     = (float*)ws;     ws += (size_t)S_LEN * U_CNT * sizeof(float);
    _Float16* A_h     = (_Float16*)ws;  ws += (size_t)S_LEN * U_CNT * K2H * sizeof(_Float16);
    _Float16* fcW_h   = (_Float16*)ws;  ws += (size_t)VOC * K2H * sizeof(_Float16);
    (void)ws_size; (void)in_sizes; (void)n_in; (void)out_size;

    cvt_f16_kernel<<<(VOC * K2H + 255) / 256, 256, 0, stream>>>(fc_W, fcW_h, VOC * K2H);
    uls_kernel<<<(S_LEN * U_CNT) / 8, 256, 0, stream>>>(x, active, enc_emb, user_emb, uls);
    rnn_kernel<<<U_CNT / 16, 128, 0, stream>>>(x, h0, enc_emb, W_ih, W_hh, b_rnn, out_rnn);
    wavg_kernel<<<S_LEN * U_CNT, 128, 0, stream>>>(t, s, active, user_emb, uls, out_rnn, A_h);
    dim3 fgrid((S_LEN * U_CNT) / 128, (VOC + 127) / 128);
    fc_kernel<<<fgrid, 256, 0, stream>>>(A_h, fcW_h, fc_b, y);
}
